// AttentionHead_72138270703923
// MI455X (gfx1250) — matrix operand order repaired and checked by reading, not compiled, not run on hardware
//
#include <hip/hip_runtime.h>

// ---------------------------------------------------------------------------
// GAT attention head for MI455X (gfx1250, wave32, WMMA).
// B=50000, N=32, D=U=128. Main GEMM done as bf16x3-split WMMA (fp32-accurate),
// logits/softmax via algebraic folding wn = W@a_neigh, wa = W@a_self.
// ---------------------------------------------------------------------------

typedef __attribute__((ext_vector_type(16))) __bf16 v16bf;
typedef __attribute__((ext_vector_type(8)))  float  v8f;
typedef __attribute__((ext_vector_type(8)))  unsigned int v8u;

#define B_TOTAL 50000
#define NNEIGH  32
#define DIM     128
#define LRELU   0.2f

__device__ __forceinline__ unsigned f2bf(float x) {
  unsigned u = __float_as_uint(x);
  return (u + 0x7FFFu + ((u >> 16) & 1u)) >> 16;   // RTNE
}
__device__ __forceinline__ float bf2f(unsigned h) {
  return __uint_as_float(h << 16);
}
// Split two fp32 into packed bf16 {hi,hi} and residual {lo,lo} dwords.
__device__ __forceinline__ uint2 split2(float x0, float x1) {
  unsigned h0 = f2bf(x0), h1 = f2bf(x1);
  unsigned l0 = f2bf(x0 - bf2f(h0)), l1 = f2bf(x1 - bf2f(h1));
  return make_uint2(h0 | (h1 << 16), l0 | (l1 << 16));
}

// ---------------------------------------------------------------------------
// Pre-kernel: wa[d] = sum_u W[d][u]*a_self[u], wn[d] = sum_u W[d][u]*a_neigh[u]
// ws[0..127] = wa, ws[128..255] = wn
// ---------------------------------------------------------------------------
__global__ void gat_precompute(const float* __restrict__ W,
                               const float* __restrict__ a_self,
                               const float* __restrict__ a_neigh,
                               float* __restrict__ ws) {
  int d = threadIdx.x;
  float s1 = 0.f, s2 = 0.f;
  for (int u = 0; u < DIM; ++u) {
    float w = W[d * DIM + u];
    s1 += w * a_self[u];
    s2 += w * a_neigh[u];
  }
  ws[d]       = s1;
  ws[DIM + d] = s2;
}

// ---------------------------------------------------------------------------
// Main kernel: 256 threads (8 waves). Wave w owns output columns [16w,16w+16).
// ---------------------------------------------------------------------------
__global__ __launch_bounds__(256)
void gat_kernel(const float* __restrict__ self_vecs,
                const float* __restrict__ neigh,
                const float* __restrict__ W,
                const float* __restrict__ wsv,
                float* __restrict__ out) {
  // LDS: A-matrix fragments in WMMA layout (bf16 hi & lo), logits, attend.
  __shared__ __align__(16) unsigned sAhi[2 * 4 * 32 * 8];  // 8 KB
  __shared__ __align__(16) unsigned sAlo[2 * 4 * 32 * 8];  // 8 KB
  __shared__ __align__(16) float sLogits[NNEIGH];
  __shared__ __align__(16) float sAttend[NNEIGH];
  __shared__ float sSelf;

  const int tid     = threadIdx.x;
  const int wid     = tid >> 5;          // wave 0..7
  const int lane    = tid & 31;
  const int colLane = lane & 15;
  const int half    = lane >> 4;         // 0: lanes 0-15, 1: lanes 16-31

  // --- Preload W as per-wave B fragments (bf16 hi/lo), kept in registers ---
  // B layout (K x N, 16x16x32): lanes 0-15 col=lane hold K {0..7, 16..23},
  // lanes 16-31 hold K {8..15, 24..31}.
  v16bf bhi[4], blo[4];
  {
    const int col = wid * 16 + colLane;
    #pragma unroll
    for (int kc = 0; kc < 4; ++kc) {
      v8u h, l;
      #pragma unroll
      for (int j = 0; j < 8; ++j) {
        int k = kc * 32 + half * 8 + (j >> 2) * 16 + 2 * (j & 3);
        float x0 = W[(size_t)k * DIM + col];
        float x1 = W[(size_t)(k + 1) * DIM + col];
        uint2 t = split2(x0, x1);
        h[j] = t.x;
        l[j] = t.y;
      }
      bhi[kc] = __builtin_bit_cast(v16bf, h);
      blo[kc] = __builtin_bit_cast(v16bf, l);
    }
  }

  // --- Phase-A role of this thread: (mtile, kchunk, lane) -> one A fragment ---
  const int tMtile = tid >> 7;                  // 0..1
  const int tK     = (tid >> 5) & 3;            // 0..3
  const int m_row  = tMtile * 16 + colLane;     // neighbor row 0..31
  // A layout (M x K, 16x32 bf16): lanes 0-15 hold K {b..b+7, b+16..b+23} with
  // b = 0; lanes 16-31 with b = 8.  idx 0..7 <-> K kA..kA+7, idx 8..15 <-> +16.
  const int kA = tK * 32 + half * 8;
  const int kB = kA + 16;

  // wn slice for this thread's K elements (constant across b).
  const float* wn = wsv + DIM;
  const float4 wn0 = *(const float4*)(wn + kA);
  const float4 wn1 = *(const float4*)(wn + kA + 4);
  const float4 wn2 = *(const float4*)(wn + kB);
  const float4 wn3 = *(const float4*)(wn + kB + 4);

  // wa slice (wave 0 computes self_f1).
  float4 wa4 = make_float4(0.f, 0.f, 0.f, 0.f);
  if (wid == 0) wa4 = *(const float4*)(wsv + 4 * lane);

  if (tid < NNEIGH) sLogits[tid] = 0.f;
  __syncthreads();

  for (int b = blockIdx.x; b < B_TOTAL; b += gridDim.x) {
    // ================= Phase A: stage + logits partials =================
    const float* rowp = neigh + ((size_t)b * NNEIGH + m_row) * DIM;
    float4 x0 = *(const float4*)(rowp + kA);
    float4 x1 = *(const float4*)(rowp + kA + 4);
    float4 x2 = *(const float4*)(rowp + kB);
    float4 x3 = *(const float4*)(rowp + kB + 4);

    // Prefetch next iteration's tile (global_prefetch_b8).
    {
      int bn = b + gridDim.x;
      if (bn < B_TOTAL)
        __builtin_prefetch(neigh + ((size_t)bn * NNEIGH + m_row) * DIM + kA, 0, 1);
    }

    // logits partial: exact fp32 dot with wn
    float p = x0.x * wn0.x + x0.y * wn0.y + x0.z * wn0.z + x0.w * wn0.w
            + x1.x * wn1.x + x1.y * wn1.y + x1.z * wn1.z + x1.w * wn1.w
            + x2.x * wn2.x + x2.y * wn2.y + x2.z * wn2.z + x2.w * wn2.w
            + x3.x * wn3.x + x3.y * wn3.y + x3.z * wn3.z + x3.w * wn3.w;
    p += __shfl_xor(p, 16, 32);                 // combine K-halves (same row)
    if (half == 0) atomicAdd(&sLogits[m_row], p);  // ds_add_f32, 4 adds/row

    // bf16 hi/lo split, stored in fragment order
    v8u hv, lv;
    {
      uint2 t;
      t = split2(x0.x, x0.y); hv[0] = t.x; lv[0] = t.y;
      t = split2(x0.z, x0.w); hv[1] = t.x; lv[1] = t.y;
      t = split2(x1.x, x1.y); hv[2] = t.x; lv[2] = t.y;
      t = split2(x1.z, x1.w); hv[3] = t.x; lv[3] = t.y;
      t = split2(x2.x, x2.y); hv[4] = t.x; lv[4] = t.y;
      t = split2(x2.z, x2.w); hv[5] = t.x; lv[5] = t.y;
      t = split2(x3.x, x3.y); hv[6] = t.x; lv[6] = t.y;
      t = split2(x3.z, x3.w); hv[7] = t.x; lv[7] = t.y;
    }
    const int frag = ((tMtile * 4 + tK) * 32 + lane) * 8;
    *(uint4*)&sAhi[frag]     = make_uint4(hv[0], hv[1], hv[2], hv[3]);
    *(uint4*)&sAhi[frag + 4] = make_uint4(hv[4], hv[5], hv[6], hv[7]);
    *(uint4*)&sAlo[frag]     = make_uint4(lv[0], lv[1], lv[2], lv[3]);
    *(uint4*)&sAlo[frag + 4] = make_uint4(lv[4], lv[5], lv[6], lv[7]);

    // self_f1 = self_vecs[b] . wa   (wave 0)
    if (wid == 0) {
      float4 sv = *(const float4*)(self_vecs + (size_t)b * DIM + 4 * lane);
      float s = sv.x * wa4.x + sv.y * wa4.y + sv.z * wa4.z + sv.w * wa4.w;
      s += __shfl_xor(s, 16, 32);
      s += __shfl_xor(s, 8, 32);
      s += __shfl_xor(s, 4, 32);
      s += __shfl_xor(s, 2, 32);
      s += __shfl_xor(s, 1, 32);
      if (lane == 0) sSelf = s;
    }
    __syncthreads();

    // ================= Phase B: bf16x3 WMMA (neigh_t tiles) =================
    v8f c0 = {};   // rows 0..15  of this wave's 16 columns
    v8f c1 = {};   // rows 16..31
    #pragma unroll
    for (int kc = 0; kc < 4; ++kc) {
      const v16bf ah0 = *(const v16bf*)&sAhi[(kc * 32 + lane) * 8];
      const v16bf al0 = *(const v16bf*)&sAlo[(kc * 32 + lane) * 8];
      const v16bf ah1 = *(const v16bf*)&sAhi[((4 + kc) * 32 + lane) * 8];
      const v16bf al1 = *(const v16bf*)&sAlo[((4 + kc) * 32 + lane) * 8];
      c0 = __builtin_amdgcn_wmma_f32_16x16x32_bf16(false, ah0, false, bhi[kc], (short)0, c0, false, false);
      c0 = __builtin_amdgcn_wmma_f32_16x16x32_bf16(false, ah0, false, blo[kc], (short)0, c0, false, false);
      c0 = __builtin_amdgcn_wmma_f32_16x16x32_bf16(false, al0, false, bhi[kc], (short)0, c0, false, false);
      c1 = __builtin_amdgcn_wmma_f32_16x16x32_bf16(false, ah1, false, bhi[kc], (short)0, c1, false, false);
      c1 = __builtin_amdgcn_wmma_f32_16x16x32_bf16(false, ah1, false, blo[kc], (short)0, c1, false, false);
      c1 = __builtin_amdgcn_wmma_f32_16x16x32_bf16(false, al1, false, bhi[kc], (short)0, c1, false, false);
    }

    // Wave 0: leaky-relu + stable softmax over 32 logits -> attend[]
    if (wid == 0) {
      float lg = sLogits[lane] + sSelf;
      float xv = lg >= 0.f ? lg : LRELU * lg;
      float mx = xv;
      mx = fmaxf(mx, __shfl_xor(mx, 16, 32));
      mx = fmaxf(mx, __shfl_xor(mx, 8, 32));
      mx = fmaxf(mx, __shfl_xor(mx, 4, 32));
      mx = fmaxf(mx, __shfl_xor(mx, 2, 32));
      mx = fmaxf(mx, __shfl_xor(mx, 1, 32));
      float e = __expf(xv - mx);
      float sum = e;
      sum += __shfl_xor(sum, 16, 32);
      sum += __shfl_xor(sum, 8, 32);
      sum += __shfl_xor(sum, 4, 32);
      sum += __shfl_xor(sum, 2, 32);
      sum += __shfl_xor(sum, 1, 32);
      sAttend[lane] = e / sum;
      sLogits[lane] = 0.f;                 // reset for next b
    }
    __syncthreads();

    // ================= Phase C: out = attend @ neigh_t, leaky-relu =========
    // C layout: c0 VGPR r -> row r (lanes 0-15) / row r+8 (lanes 16-31);
    //           c1 same +16.
    const int base0 = half * 8;
    float4 a0 = *(const float4*)&sAttend[base0];
    float4 a1 = *(const float4*)&sAttend[base0 + 4];
    float4 a2 = *(const float4*)&sAttend[16 + base0];
    float4 a3 = *(const float4*)&sAttend[16 + base0 + 4];
    float acc = a0.x * c0[0] + a0.y * c0[1] + a0.z * c0[2] + a0.w * c0[3]
              + a1.x * c0[4] + a1.y * c0[5] + a1.z * c0[6] + a1.w * c0[7]
              + a2.x * c1[0] + a2.y * c1[1] + a2.z * c1[2] + a2.w * c1[3]
              + a3.x * c1[4] + a3.y * c1[5] + a3.z * c1[6] + a3.w * c1[7];
    acc += __shfl_xor(acc, 16, 32);        // combine the two row-halves
    float r = acc >= 0.f ? acc : LRELU * acc;
    if (half == 0) out[(size_t)b * DIM + wid * 16 + colLane] = r;
    __syncthreads();                        // protect LDS reuse next iter
  }
}

// ---------------------------------------------------------------------------
extern "C" void kernel_launch(void* const* d_in, const int* in_sizes, int n_in,
                              void* d_out, int out_size, void* d_ws, size_t ws_size,
                              hipStream_t stream) {
  const float* self_vecs = (const float*)d_in[0];  // [B,128]
  const float* neigh     = (const float*)d_in[1];  // [B,32,128]
  const float* W         = (const float*)d_in[2];  // [128,128]
  const float* a_self    = (const float*)d_in[3];  // [128,1]
  const float* a_neigh   = (const float*)d_in[4];  // [128,1]
  float* out = (float*)d_out;                      // [B,128]
  float* ws  = (float*)d_ws;                       // 256 floats used

  hipLaunchKernelGGL(gat_precompute, dim3(1), dim3(128), 0, stream,
                     W, a_self, a_neigh, ws);
  hipLaunchKernelGGL(gat_kernel, dim3(2048), dim3(256), 0, stream,
                     self_vecs, neigh, W, ws, out);
}
